// _ContextAttentionPooling_1073741824708
// MI455X (gfx1250) — compile-verified, matrix operand order checked by reading, not run on hardware
//
#include <hip/hip_runtime.h>
#include <math.h>

// Problem constants (match reference)
#define B_SZ    32
#define S_SZ    4096
#define KDIM    512     // Q_DIM == KV_DIM
#define ADIM    128     // A_DIM
#define CHUNK   128     // S rows per workgroup in the main kernel
#define NCHUNK  (S_SZ / CHUNK)   // 32
#define NKSTEP  (KDIM / 32)      // 16 K-steps of 32
#define NTILE_N (ADIM / 16)      // 8  N-tiles

typedef __attribute__((ext_vector_type(16))) _Float16 v16h;
typedef __attribute__((ext_vector_type(8)))  float    v8f;

// ---------------------------------------------------------------------------
// Load one 16x32 f16 fragment (A layout; B uses the same pattern with N as the
// "row") from a row-major f32 matrix, converting to f16 on the fly.
// ISA 16-bit A layout (wave32): lane L holds row (L&15); half = L>>4 adds 8 to
// K; VGPR pair j holds K = (j<4 ? 2j : 16+2(j-4)) and K+1.
// ---------------------------------------------------------------------------
__device__ __forceinline__ v16h load_frag_f16(const float* __restrict__ base,
                                              int ld, int row0, int k0, int lane) {
  const int r    = row0 + (lane & 15);
  const int half = lane >> 4;
  const float* p = base + (size_t)r * ld + k0 + half * 8;
  v16h f;
#pragma unroll
  for (int j = 0; j < 8; ++j) {
    const int k = (j < 4) ? (2 * j) : (16 + 2 * (j - 4));
    f[2 * j]     = (_Float16)p[k];
    f[2 * j + 1] = (_Float16)p[k + 1];
  }
  return f;
}

// ---------------------------------------------------------------------------
// Kernel 1: q = query @ Wq^T   [32x512] x [128x512]^T -> [32x128]
// One workgroup, 8 waves; wave w owns N-tile w, both M-tiles (rows 0-15,16-31).
// ---------------------------------------------------------------------------
__global__ void __launch_bounds__(256)
qproj_kernel(const float* __restrict__ query, const float* __restrict__ Wq,
             float* __restrict__ qproj) {
  const int wave = threadIdx.x >> 5;   // 0..7 == N-tile
  const int lane = threadIdx.x & 31;

  v8f acc0 = {};
  v8f acc1 = {};
  for (int ks = 0; ks < NKSTEP; ++ks) {
    const int k0 = ks * 32;
    v16h bf = load_frag_f16(Wq, KDIM, wave * 16, k0, lane);     // B[k][n]=Wq[n][k]
    v16h a0 = load_frag_f16(query, KDIM, 0,  k0, lane);
    v16h a1 = load_frag_f16(query, KDIM, 16, k0, lane);
    acc0 = __builtin_amdgcn_wmma_f32_16x16x32_f16(false, a0, false, bf,
                                                  (short)0, acc0, false, false);
    acc1 = __builtin_amdgcn_wmma_f32_16x16x32_f16(false, a1, false, bf,
                                                  (short)0, acc1, false, false);
  }
  const int n    = wave * 16 + (lane & 15);
  const int half = lane >> 4;
#pragma unroll
  for (int r = 0; r < 8; ++r) {
    qproj[(size_t)(r + 8 * half) * ADIM + n]      = acc0[r];
    qproj[(size_t)(16 + r + 8 * half) * ADIM + n] = acc1[r];
  }
}

// ---------------------------------------------------------------------------
// Kernel 2: per (batch b, chunk of 128 S-rows):
//   - Wk staged once per WG into LDS as f16, pre-swizzled into WMMA fragment
//     order (one v16h LDS read per B-fragment per lane)
//   - kproj = kv_chunk @ Wk^T via v_wmma_f32_16x16x32_f16
//   - score_s = sum_a Ws[a]*tanh(q[b][a] + kproj[s][a]); mask -> -inf
//   - chunk-local softmax stats (m, l) + partial context (chunk re-read
//     hits L2, not HBM)
// LDS: the 128KB swizzled-Wk buffer and the 66KB kproj spill tile are not
// live simultaneously -> union them (barrier in between); ~130KB total so
// 2 workgroups/WGP fit in the 320KB LDS pool.
// ---------------------------------------------------------------------------
#define WK_F16_ELTS   (NKSTEP * NTILE_N * 512)             // 65536 f16 = 128KB
#define LDSK_BYTES    (CHUNK * (ADIM + 1) * 4)             // 66048 B
#define LDS_UNION_SZ  ((WK_F16_ELTS * 2 > LDSK_BYTES) ? WK_F16_ELTS * 2 : LDSK_BYTES)

__global__ void __launch_bounds__(256)
score_partial_kernel(const float* __restrict__ kv,
                     const unsigned char* __restrict__ mask,
                     const float* __restrict__ Wk,
                     const float* __restrict__ Ws,
                     const float* __restrict__ qproj,
                     float* __restrict__ scores,   // [B][S]
                     float* __restrict__ pacc,     // [B][NCHUNK][KDIM]
                     float* __restrict__ pml) {    // [B][NCHUNK][2]
  const int chunk = blockIdx.x;
  const int b     = blockIdx.y;
  const int wave  = threadIdx.x >> 5;   // 0..7 == M-tile within chunk
  const int lane  = threadIdx.x & 31;
  const int s0    = chunk * CHUNK;

  __shared__ __align__(16) unsigned char lds_raw[LDS_UNION_SZ];
  __shared__ float lds_q[ADIM];
  __shared__ float lds_w[ADIM];
  __shared__ float lds_p[CHUNK];
  __shared__ float lds_red[2];

  _Float16* lds_wk = (_Float16*)lds_raw;                      // phase 1
  float (*lds_k)[ADIM + 1] = (float(*)[ADIM + 1])lds_raw;     // phase 2

  if (threadIdx.x < ADIM) {
    lds_q[threadIdx.x] = qproj[b * ADIM + threadIdx.x];
    lds_w[threadIdx.x] = Ws[threadIdx.x];
  }

  // --- stage Wk -> LDS, f16, fragment-swizzled (4 consecutive k per iter) ---
  for (int idx = threadIdx.x; idx < (ADIM * KDIM) / 4; idx += 256) {
    const int n_row = (idx * 4) >> 9;        // 0..127 (output column a)
    const int k     = (idx * 4) & 511;       // multiple of 4
    const float4 v  = *(const float4*)(Wk + (size_t)n_row * KDIM + k);
    const int ks = k >> 5;
    const int kk = k & 31;
    int half, e;
    if (kk < 16) { half = kk >> 3;        e = kk & 7; }
    else         { half = (kk - 16) >> 3; e = 8 + ((kk - 16) & 7); }
    const int l  = half * 16 + (n_row & 15);     // lane owning this run
    const int nt = n_row >> 4;                   // N-tile
    _Float16* dst = &lds_wk[(size_t)(((ks * NTILE_N + nt) * 32 + l) * 16 + e)];
    dst[0] = (_Float16)v.x;
    dst[1] = (_Float16)v.y;
    dst[2] = (_Float16)v.z;
    dst[3] = (_Float16)v.w;
  }
  __syncthreads();

  const float* kvb = kv + (size_t)b * S_SZ * KDIM;

  // --- projection GEMM: 128 x 512 x 128, wave owns M-tile `wave` ---
  v8f acc[NTILE_N] = {};
  for (int ks = 0; ks < NKSTEP; ++ks) {
    const int k0 = ks * 32;
    const v16h af = load_frag_f16(kvb, KDIM, s0 + wave * 16, k0, lane);
#pragma unroll
    for (int n = 0; n < NTILE_N; ++n) {
      const v16h bf =
          *(const v16h*)&lds_wk[(size_t)(((ks * NTILE_N + n) * 32 + lane) * 16)];
      acc[n] = __builtin_amdgcn_wmma_f32_16x16x32_f16(false, af, false, bf,
                                                      (short)0, acc[n], false, false);
    }
  }
  __syncthreads();   // lds_wk dead; lds_k aliases the same storage

  // spill accumulators (f32 C layout) to LDS
  {
    const int n16 = lane & 15, half = lane >> 4;
#pragma unroll
    for (int n = 0; n < NTILE_N; ++n)
#pragma unroll
      for (int r = 0; r < 8; ++r)
        lds_k[wave * 16 + r + 8 * half][n * 16 + n16] = acc[n][r];
  }
  __syncthreads();

  // --- scores: threads 0..127 each own one S-row of the chunk ---
  if (threadIdx.x < CHUNK) {
    const int s = threadIdx.x;
    float sum = 0.0f;
#pragma unroll 4
    for (int a = 0; a < ADIM; ++a)
      sum += lds_w[a] * tanhf(lds_q[a] + lds_k[s][a]);
    const int sg = s0 + s;
    const bool mk = mask[(size_t)b * S_SZ + sg] != 0;
    const float sc = mk ? sum : -INFINITY;
    scores[(size_t)b * S_SZ + sg] = sc;
    lds_p[s] = sc;
  }
  __syncthreads();

  if (threadIdx.x == 0) {
    float m = -INFINITY;
    for (int s = 0; s < CHUNK; ++s) m = fmaxf(m, lds_p[s]);
    lds_red[0] = m;
  }
  __syncthreads();
  const float m = lds_red[0];

  if (threadIdx.x < CHUNK) {
    const float sc = lds_p[threadIdx.x];
    lds_p[threadIdx.x] =
        (m > -INFINITY && sc > -INFINITY) ? __expf(sc - m) : 0.0f;
  }
  __syncthreads();

  if (threadIdx.x == 0) {
    float l = 0.0f;
    for (int s = 0; s < CHUNK; ++s) l += lds_p[s];
    pml[((size_t)b * NCHUNK + chunk) * 2 + 0] = m;
    pml[((size_t)b * NCHUNK + chunk) * 2 + 1] = l;
  }

  // --- partial context: thread t owns columns 2t, 2t+1 (L2-resident reread) ---
  float a0 = 0.0f, a1 = 0.0f;
  const int c = threadIdx.x * 2;
  for (int s = 0; s < CHUNK; ++s) {
    const float p = lds_p[s];
    if (p == 0.0f) continue;
    const float* row = kvb + (size_t)(s0 + s) * KDIM;
    a0 += p * row[c];
    a1 += p * row[c + 1];
  }
  float* pa = pacc + ((size_t)b * NCHUNK + chunk) * KDIM;
  pa[c]     = a0;
  pa[c + 1] = a1;
}

// ---------------------------------------------------------------------------
// Kernel 3: per batch: merge chunk partials (global softmax) -> context +
// normalized weights. out = [context (32x512) | weights (32x4096)]
// ---------------------------------------------------------------------------
__global__ void __launch_bounds__(256)
combine_kernel(const float* __restrict__ scores,
               const float* __restrict__ pacc,
               const float* __restrict__ pml,
               float* __restrict__ out) {
  const int b = blockIdx.x;
  const int t = threadIdx.x;

  __shared__ float sM, sL;
  __shared__ float scale[NCHUNK];

  if (t == 0) {
    float M = -INFINITY;
    for (int c = 0; c < NCHUNK; ++c)
      M = fmaxf(M, pml[((size_t)b * NCHUNK + c) * 2 + 0]);
    float L = 0.0f;
    for (int c = 0; c < NCHUNK; ++c) {
      const float mc = pml[((size_t)b * NCHUNK + c) * 2 + 0];
      const float lc = pml[((size_t)b * NCHUNK + c) * 2 + 1];
      const float sc = (mc > -INFINITY && M > -INFINITY) ? expf(mc - M) : 0.0f;
      scale[c] = sc;
      L += lc * sc;
    }
    sM = M;
    sL = L;
  }
  __syncthreads();

  const float M    = sM;
  const float invL = (sL > 0.0f) ? 1.0f / sL : 0.0f;

  // context
  for (int c = t; c < KDIM; c += 256) {
    float a = 0.0f;
#pragma unroll 4
    for (int ch = 0; ch < NCHUNK; ++ch)
      a += pacc[((size_t)b * NCHUNK + ch) * KDIM + c] * scale[ch];
    out[(size_t)b * KDIM + c] = a * invL;
  }

  // weights
  float* wout = out + (size_t)B_SZ * KDIM;
  for (int s = t; s < S_SZ; s += 256) {
    const float sc = scores[(size_t)b * S_SZ + s];
    const float w =
        (M > -INFINITY && sc > -INFINITY) ? expf(sc - M) * invL : 0.0f;
    wout[(size_t)b * S_SZ + s] = w;
  }
}

// ---------------------------------------------------------------------------
extern "C" void kernel_launch(void* const* d_in, const int* in_sizes, int n_in,
                              void* d_out, int out_size, void* d_ws, size_t ws_size,
                              hipStream_t stream) {
  const float*         query = (const float*)d_in[0];
  const float*         kv    = (const float*)d_in[1];
  const unsigned char* mask  = (const unsigned char*)d_in[2];
  const float*         Wq    = (const float*)d_in[3];
  const float*         Wk    = (const float*)d_in[4];
  const float*         Ws    = (const float*)d_in[5];
  float*               out   = (float*)d_out;

  // workspace layout (floats)
  float* ws     = (float*)d_ws;
  float* qproj  = ws;                                   // B*ADIM         = 4096
  float* scores = qproj + (size_t)B_SZ * ADIM;          // B*S            = 131072
  float* pacc   = scores + (size_t)B_SZ * S_SZ;         // B*NCHUNK*KDIM  = 524288
  float* pml    = pacc + (size_t)B_SZ * NCHUNK * KDIM;  // B*NCHUNK*2     = 2048

  qproj_kernel<<<1, 256, 0, stream>>>(query, Wq, qproj);
  score_partial_kernel<<<dim3(NCHUNK, B_SZ), 256, 0, stream>>>(
      kv, mask, Wk, Ws, qproj, scores, pacc, pml);
  combine_kernel<<<B_SZ, 256, 0, stream>>>(scores, pacc, pml, out);
}
